// Attention_36146444763313
// MI455X (gfx1250) — matrix operand order repaired and checked by reading, not compiled, not run on hardware
//
#include <hip/hip_runtime.h>
#include <hip/hip_fp16.h>

#define DIM      2048
#define HEAD_DIM 128
#define N_HEADS  16
#define N_KV     4
#define BATCH    2
#define SEQ      2048
#define QKV_N    ((N_HEADS + 2 * N_KV) * HEAD_DIM)   // 3072
#define ROWS     (BATCH * SEQ)                       // 4096

typedef __attribute__((ext_vector_type(16))) _Float16 v16h;
typedef __attribute__((ext_vector_type(8)))  _Float16 v8h;
typedef __attribute__((ext_vector_type(8)))  float    v8f;
typedef __attribute__((ext_vector_type(4)))  int      v4i_t;

union V16 { v16h v; struct { v8h lo, hi; } s; };

#if __has_builtin(__builtin_amdgcn_global_load_async_to_lds_b128) && \
    __has_builtin(__builtin_amdgcn_s_wait_asynccnt)
#define HAVE_ASYNC_LDS 1
typedef __attribute__((address_space(1))) v4i_t* gp4;   // global int4*
typedef __attribute__((address_space(3))) v4i_t* lp4;   // LDS int4*
#else
#define HAVE_ASYNC_LDS 0
#endif

// ---------------------------------------------------------------------------
// 1) Per-row abs-max quantization of x: store round(x*s) as f16 (exact ints),
//    rowscale = maxabs/127 (dequant factor folded into GEMM epilogue).
// ---------------------------------------------------------------------------
__global__ void quant_rows_k(const float* __restrict__ x, _Float16* __restrict__ q,
                             float* __restrict__ rowscale, int K) {
  const int row = blockIdx.x;
  const float* xr = x + (size_t)row * K;
  __shared__ float red[256];
  float m = 0.f;
  for (int c = threadIdx.x; c < K; c += 256) m = fmaxf(m, fabsf(xr[c]));
  red[threadIdx.x] = m;
  __syncthreads();
  for (int o = 128; o > 0; o >>= 1) {
    if (threadIdx.x < o) red[threadIdx.x] = fmaxf(red[threadIdx.x], red[threadIdx.x + o]);
    __syncthreads();
  }
  const float ma = fmaxf(red[0], 1e-5f);
  const float s = 127.f / ma;
  _Float16* qr = q + (size_t)row * K;
  for (int c = threadIdx.x; c < K; c += 256) {
    float v = rintf(xr[c] * s);
    v = fminf(fmaxf(v, -128.f), 127.f);
    qr[c] = (_Float16)v;
  }
  if (threadIdx.x == 0) rowscale[row] = ma / 127.f;
}

// ---------------------------------------------------------------------------
// 2) f32 -> f16 weight conversion
// ---------------------------------------------------------------------------
__global__ void cvt_f16_k(const float* __restrict__ a, _Float16* __restrict__ o, size_t n) {
  size_t i = (size_t)blockIdx.x * blockDim.x + threadIdx.x;
  if (i < n) o[i] = (_Float16)a[i];
}

// ---------------------------------------------------------------------------
// 3) WMMA GEMM: out[r][n] = rowscale[r] * sum_k A16[r][k] * W16[n][k]
//    Each wave computes a 16x64 strip: one A fragment feeds 4 WMMAs.
// ---------------------------------------------------------------------------
__global__ void gemm_wmma_k(const _Float16* __restrict__ A, const _Float16* __restrict__ W,
                            const float* __restrict__ rowscale, float* __restrict__ out,
                            int N, int K) {
  const int lane = threadIdx.x & 31;
  const int wave = threadIdx.x >> 5;
  const int n0 = (blockIdx.x * 4 + wave) * 64;
  const int r0 = blockIdx.y * 16;
  const int hf  = lane >> 4;
  const int l16 = lane & 15;

  const _Float16* aRow  = A + (size_t)(r0 + l16) * K + hf * 16;
  const _Float16* bRow0 = W + (size_t)(n0 + l16) * K + hf * 16;

  v8f acc[4];
  {
    v8f z = {};
    for (int t = 0; t < 4; ++t) acc[t] = z;
  }
  for (int k0 = 0; k0 < K; k0 += 32) {
    __builtin_prefetch(aRow + k0 + 256, 0, 1);
    V16 a;
    a.s.lo = *(const v8h*)(aRow + k0);
    a.s.hi = *(const v8h*)(aRow + k0 + 8);
    for (int t = 0; t < 4; ++t) {
      const _Float16* bp = bRow0 + (size_t)t * 16 * K + k0;
      __builtin_prefetch(bp + 256, 0, 1);
      V16 b;
      b.s.lo = *(const v8h*)(bp);
      b.s.hi = *(const v8h*)(bp + 8);
      acc[t] = __builtin_amdgcn_wmma_f32_16x16x32_f16(false, a.v, false, b.v,
                                                      (short)0, acc[t], false, false);
    }
  }
  for (int t = 0; t < 4; ++t) {
    const int n = n0 + 16 * t + l16;
    for (int r = 0; r < 8; ++r) {
      const int row = r0 + r + hf * 8;            // C layout: VGPR r -> M=r (+8 hi half)
      out[(size_t)row * N + n] = acc[t][r] * rowscale[row];
    }
  }
}

// ---------------------------------------------------------------------------
// 4) RoPE + head split.  xqkv f32 [ROWS][3072] ->
//    q16 [b][h][s][128] (rotated), k16 [b][g][s][128] (rotated),
//    vT16 [b][g][128][SEQ] (transposed so attention tiles are contiguous).
// ---------------------------------------------------------------------------
__global__ void rope_split_k(const float* __restrict__ xqkv,
                             _Float16* __restrict__ q, _Float16* __restrict__ k,
                             _Float16* __restrict__ vT) {
  const int row = blockIdx.y;                         // b*SEQ + s
  const int c2 = blockIdx.x * blockDim.x + threadIdx.x;   // pair index 0..1535
  const int b = row >> 11, s = row & (SEQ - 1);
  const float* xr = xqkv + (size_t)row * QKV_N;

  if (c2 < 1024) {                                    // Q: 16 heads * 64 pairs
    const int h = c2 >> 6, i = c2 & 63;
    const float fr = powf(500000.0f, -(float)(2 * i) / 128.0f);
    float sn, cs; __sincosf((float)s * fr, &sn, &cs);
    const float x0 = xr[h * 128 + 2 * i], x1 = xr[h * 128 + 2 * i + 1];
    _Float16* dst = q + (((size_t)(b * N_HEADS + h)) * SEQ + s) * HEAD_DIM + 2 * i;
    dst[0] = (_Float16)(x0 * cs - x1 * sn);
    dst[1] = (_Float16)(x0 * sn + x1 * cs);
  } else if (c2 < 1280) {                             // K: 4 groups * 64 pairs
    const int p = c2 - 1024, g = p >> 6, i = p & 63;
    const float fr = powf(500000.0f, -(float)(2 * i) / 128.0f);
    float sn, cs; __sincosf((float)s * fr, &sn, &cs);
    const float x0 = xr[2048 + g * 128 + 2 * i], x1 = xr[2048 + g * 128 + 2 * i + 1];
    _Float16* dst = k + (((size_t)(b * N_KV + g)) * SEQ + s) * HEAD_DIM + 2 * i;
    dst[0] = (_Float16)(x0 * cs - x1 * sn);
    dst[1] = (_Float16)(x0 * sn + x1 * cs);
  } else {                                            // V: transpose store
    const int p = c2 - 1280, g = p >> 6, i = p & 63;
    const float x0 = xr[2560 + g * 128 + 2 * i], x1 = xr[2560 + g * 128 + 2 * i + 1];
    _Float16* base = vT + (((size_t)(b * N_KV + g)) * HEAD_DIM + 2 * i) * SEQ + s;
    base[0]   = (_Float16)x0;
    base[SEQ] = (_Float16)x1;
  }
}

// ---------------------------------------------------------------------------
// 5) Flash attention with async double-buffered K/V tiles in LDS.
//    Tile j+1 is DMA'd via GLOBAL_LOAD_ASYNC_TO_LDS_B128 while WMMAs consume
//    tile j; ASYNCcnt completes in order, so waiting <= 32 retires exactly the
//    previous tile's 32 transfers.
// ---------------------------------------------------------------------------
__device__ __forceinline__ void issue_tile_copy(const _Float16* __restrict__ kbase,
                                                const _Float16* __restrict__ vbase,
                                                int kb, _Float16* kt, _Float16* vt,
                                                int lane) {
#if HAVE_ASYNC_LDS
  // K tile: 32 rows x 128 halves = 512 x 16B transfers, 16 per lane
  for (int i = 0; i < 16; ++i) {
    const int idx = lane + 32 * i;                    // 0..511
    const int row = idx >> 4, seg = idx & 15;
    const _Float16* g = kbase + (size_t)(kb + row) * HEAD_DIM + seg * 8;
    __builtin_amdgcn_global_load_async_to_lds_b128(
        (gp4)(v4i_t*)(g), (lp4)(v4i_t*)(kt + idx * 8), 0, 0);
  }
  // V tile (from vT): 128 rows x 32 halves = 512 x 16B transfers, 16 per lane
  for (int i = 0; i < 16; ++i) {
    const int idx = lane + 32 * i;                    // 0..511
    const int d = idx >> 2, seg = idx & 3;
    const _Float16* g = vbase + (size_t)d * SEQ + kb + seg * 8;
    __builtin_amdgcn_global_load_async_to_lds_b128(
        (gp4)(v4i_t*)(g), (lp4)(v4i_t*)(vt + idx * 8), 0, 0);
  }
#else
  for (int i = 0; i < 16; ++i) {
    const int idx = lane + 32 * i;
    const int row = idx >> 4, seg = idx & 15;
    *(v8h*)(kt + idx * 8) = *(const v8h*)(kbase + (size_t)(kb + row) * HEAD_DIM + seg * 8);
  }
  for (int i = 0; i < 16; ++i) {
    const int idx = lane + 32 * i;
    const int d = idx >> 2, seg = idx & 3;
    *(v8h*)(vt + idx * 8) = *(const v8h*)(vbase + (size_t)d * SEQ + kb + seg * 8);
  }
#endif
}

__global__ void flash_attn_k(const _Float16* __restrict__ Q, const _Float16* __restrict__ Kc,
                             const _Float16* __restrict__ Vt, float* __restrict__ outp) {
  const int lane = threadIdx.x;
  const int h = blockIdx.y, b = blockIdx.z;
  const int g = h >> 2;                               // GQA group (hpg = 4)
  const int r0 = blockIdx.x * 16;
  const int hf = lane >> 4, l16 = lane & 15;

  const _Float16* qbase = Q  + (((size_t)(b * N_HEADS + h)) * SEQ + (r0 + l16)) * HEAD_DIM + hf * 16;
  const _Float16* kbase = Kc + ((size_t)(b * N_KV + g)) * SEQ * HEAD_DIM;
  const _Float16* vbase = Vt + ((size_t)(b * N_KV + g)) * HEAD_DIM * SEQ;

  // Q tile (16x128) in A-layout: 4 chunks of K=32
  V16 qa[4];
  for (int i = 0; i < 4; ++i) {
    qa[i].s.lo = *(const v8h*)(qbase + 32 * i);
    qa[i].s.hi = *(const v8h*)(qbase + 32 * i + 8);
  }

  __shared__ _Float16 Kt[2][32 * 128];                // 2 x 8 KB
  __shared__ _Float16 Vl[2][128 * 32];                // 2 x 8 KB
  __shared__ _Float16 Pl[16 * 32];                    // 1 KB

  v8f O[8];
  {
    v8f z = {};
    for (int t = 0; t < 8; ++t) O[t] = z;
  }
  float mrow[8], lrow[8];
  for (int r = 0; r < 8; ++r) { mrow[r] = -1e30f; lrow[r] = 0.f; }

  const float sc = 0.08838834764831845f;              // 1/sqrt(128)
  const int jmax = (r0 + 15) >> 5;

  issue_tile_copy(kbase, vbase, 0, &Kt[0][0], &Vl[0][0], lane);

  for (int j = 0; j <= jmax; ++j) {
    const int kb = j * 32;
    const int buf = j & 1;
    if (j < jmax)
      issue_tile_copy(kbase, vbase, kb + 32, &Kt[buf ^ 1][0], &Vl[buf ^ 1][0], lane);
#if HAVE_ASYNC_LDS
    if (j < jmax) __builtin_amdgcn_s_wait_asynccnt(32);   // tile j's 32 transfers done
    else          __builtin_amdgcn_s_wait_asynccnt(0);
    __asm__ __volatile__("" ::: "memory");
#endif

    v8f S0 = {}, S1 = {};
    for (int i = 0; i < 4; ++i) {
      V16 b0, b1;
      const _Float16* kp0 = &Kt[buf][(size_t)l16 * HEAD_DIM + 32 * i + hf * 16];
      const _Float16* kp1 = &Kt[buf][(size_t)(16 + l16) * HEAD_DIM + 32 * i + hf * 16];
      b0.s.lo = *(const v8h*)(kp0); b0.s.hi = *(const v8h*)(kp0 + 8);
      b1.s.lo = *(const v8h*)(kp1); b1.s.hi = *(const v8h*)(kp1 + 8);
      S0 = __builtin_amdgcn_wmma_f32_16x16x32_f16(false, qa[i].v, false, b0.v, (short)0, S0, false, false);
      S1 = __builtin_amdgcn_wmma_f32_16x16x32_f16(false, qa[i].v, false, b1.v, (short)0, S1, false, false);
    }

    // online softmax per row (row elements live across a 16-lane half)
    for (int r = 0; r < 8; ++r) {
      const int m = r0 + r + hf * 8;
      float s0 = S0[r] * sc, s1 = S1[r] * sc;
      if (kb + l16 > m)      s0 = -1e30f;             // causal mask
      if (kb + 16 + l16 > m) s1 = -1e30f;
      float t = fmaxf(s0, s1);
      for (int msk = 8; msk; msk >>= 1) t = fmaxf(t, __shfl_xor(t, msk, 16));
      const float newm = fmaxf(mrow[r], t);
      const float alpha = __expf(mrow[r] - newm);
      const float p0 = __expf(s0 - newm), p1 = __expf(s1 - newm);
      float rs = p0 + p1;
      for (int msk = 8; msk; msk >>= 1) rs += __shfl_xor(rs, msk, 16);
      lrow[r] = lrow[r] * alpha + rs;
      mrow[r] = newm;
      for (int t2 = 0; t2 < 8; ++t2) O[t2][r] *= alpha;
      const int ml = r + hf * 8;
      Pl[ml * 32 + l16]      = (_Float16)p0;
      Pl[ml * 32 + 16 + l16] = (_Float16)p1;
    }
    __syncthreads();                                  // 1-wave WG: orders LDS P
    V16 pa;                                           // P (16x32) in A-layout
    pa.s.lo = *(const v8h*)(&Pl[l16 * 32 + hf * 16]);
    pa.s.hi = *(const v8h*)(&Pl[l16 * 32 + hf * 16 + 8]);
    for (int t = 0; t < 8; ++t) {
      V16 vb;
      const _Float16* vp = &Vl[buf][(size_t)(16 * t + l16) * 32 + hf * 16];
      vb.s.lo = *(const v8h*)(vp);
      vb.s.hi = *(const v8h*)(vp + 8);
      O[t] = __builtin_amdgcn_wmma_f32_16x16x32_f16(false, pa.v, false, vb.v, (short)0, O[t], false, false);
    }
    __syncthreads();
  }

  for (int r = 0; r < 8; ++r) {
    const int srow = r0 + r + hf * 8;
    const float inv = 1.f / lrow[r];
    float* orow = outp + ((size_t)b * SEQ + srow) * (N_HEADS * HEAD_DIM) + h * HEAD_DIM;
    for (int t = 0; t < 8; ++t) orow[16 * t + l16] = O[t][r] * inv;
  }
}

// ---------------------------------------------------------------------------
// 6) RMSNorm + quantize (per row of attention output)
// ---------------------------------------------------------------------------
__global__ void rmsnorm_quant_k(const float* __restrict__ a, const float* __restrict__ nw,
                                _Float16* __restrict__ q, float* __restrict__ rowscale) {
  const int row = blockIdx.x;
  const float* ar = a + (size_t)row * DIM;
  __shared__ float ybuf[DIM];
  __shared__ float red[256];
  float ss = 0.f;
  for (int c = threadIdx.x; c < DIM; c += 256) { const float v = ar[c]; ss += v * v; }
  red[threadIdx.x] = ss;
  __syncthreads();
  for (int o = 128; o > 0; o >>= 1) {
    if (threadIdx.x < o) red[threadIdx.x] += red[threadIdx.x + o];
    __syncthreads();
  }
  const float rms = rsqrtf(red[0] / (float)DIM + 1e-5f);
  __syncthreads();
  float mx = 0.f;
  for (int c = threadIdx.x; c < DIM; c += 256) {
    const float y = ar[c] * rms * nw[c];
    ybuf[c] = y;
    mx = fmaxf(mx, fabsf(y));
  }
  red[threadIdx.x] = mx;
  __syncthreads();
  for (int o = 128; o > 0; o >>= 1) {
    if (threadIdx.x < o) red[threadIdx.x] = fmaxf(red[threadIdx.x], red[threadIdx.x + o]);
    __syncthreads();
  }
  const float ma = fmaxf(red[0], 1e-5f);
  const float s = 127.f / ma;
  _Float16* qr = q + (size_t)row * DIM;
  for (int c = threadIdx.x; c < DIM; c += 256) {
    float v = rintf(ybuf[c] * s);
    v = fminf(fmaxf(v, -128.f), 127.f);
    qr[c] = (_Float16)v;
  }
  if (threadIdx.x == 0) rowscale[row] = ma / 127.f;
}

// ---------------------------------------------------------------------------
// Host launcher
// ---------------------------------------------------------------------------
extern "C" void kernel_launch(void* const* d_in, const int* in_sizes, int n_in,
                              void* d_out, int out_size, void* d_ws, size_t ws_size,
                              hipStream_t stream) {
  const float* x      = (const float*)d_in[0];
  const float* wqkv   = (const float*)d_in[3];
  const float* wo     = (const float*)d_in[4];
  const float* norm_w = (const float*)d_in[5];

  char* ws = (char*)d_ws;
  size_t off = 0;
  _Float16* xq16   = (_Float16*)(ws + off); off += (size_t)ROWS * DIM * 2;       // 16 MiB
  float*    rs1    = (float*)(ws + off);    off += (size_t)ROWS * 4;
  _Float16* wqkv16 = (_Float16*)(ws + off); off += (size_t)QKV_N * DIM * 2;      // 12 MiB
  _Float16* wo16   = (_Float16*)(ws + off); off += (size_t)DIM * DIM * 2;        // 8 MiB
  float*    xqkv   = (float*)(ws + off);    off += (size_t)ROWS * QKV_N * 4;     // 48 MiB
  _Float16* q16    = (_Float16*)(ws + off); off += (size_t)BATCH * N_HEADS * SEQ * HEAD_DIM * 2;
  _Float16* k16    = (_Float16*)(ws + off); off += (size_t)BATCH * N_KV * SEQ * HEAD_DIM * 2;
  _Float16* vT16   = (_Float16*)(ws + off); off += (size_t)BATCH * N_KV * HEAD_DIM * SEQ * 2;
  float*    attn   = (float*)(ws + off);    off += (size_t)ROWS * DIM * 4;       // 32 MiB
  _Float16* y16    = (_Float16*)(ws + off); off += (size_t)ROWS * DIM * 2;
  float*    rs2    = (float*)(ws + off);    off += (size_t)ROWS * 4;
  (void)ws_size; (void)in_sizes; (void)n_in; (void)out_size;

  // 1) quantize activations
  quant_rows_k<<<ROWS, 256, 0, stream>>>(x, xq16, rs1, DIM);

  // 2) convert weights to f16
  {
    const size_t nq = (size_t)QKV_N * DIM;
    cvt_f16_k<<<(unsigned)((nq + 255) / 256), 256, 0, stream>>>(wqkv, wqkv16, nq);
    const size_t no = (size_t)DIM * DIM;
    cvt_f16_k<<<(unsigned)((no + 255) / 256), 256, 0, stream>>>(wo, wo16, no);
  }

  // 3) GEMM1: xqkv = quant(x) @ wqkv^T  (scaled f32 epilogue)
  gemm_wmma_k<<<dim3(QKV_N / 256, ROWS / 16), 128, 0, stream>>>(xq16, wqkv16, rs1, xqkv, QKV_N, DIM);

  // 4) RoPE + split into per-head layouts (V transposed)
  rope_split_k<<<dim3(QKV_N / 2 / 256, ROWS), 256, 0, stream>>>(xqkv, q16, k16, vT16);

  // 5) causal flash attention (async LDS double-buffered K/V)
  flash_attn_k<<<dim3(SEQ / 16, N_HEADS, BATCH), 32, 0, stream>>>(q16, k16, vT16, attn);

  // 6) RMSNorm + quantize
  rmsnorm_quant_k<<<ROWS, 256, 0, stream>>>(attn, norm_w, y16, rs2);

  // 7) GEMM2: out = quant(normed) @ wo^T
  gemm_wmma_k<<<dim3(DIM / 256, ROWS / 16), 128, 0, stream>>>(y16, wo16, rs2, (float*)d_out, DIM, DIM);
}
